// DetrDecoderLayer_50775103373815
// MI455X (gfx1250) — matrix operand order repaired and checked by reading, compiled, not run on hardware
//
#include <hip/hip_runtime.h>
#include <hip/hip_bf16.h>

#define NP   512
#define MNP  4096
#define BSZ  8
#define DIM  256
#define NH   8
#define HD   32
#define TOPK 32
#define FF   2048

typedef __attribute__((ext_vector_type(16))) __bf16 v16bf;
typedef __attribute__((ext_vector_type(8)))  float  v8f;
typedef __attribute__((ext_vector_type(4)))  int    v4i;

// -------------------------------------------------------------------------
// 16-byte global -> LDS copy: async (ASYNCcnt) path on gfx1250 toolchains
// that expose the builtin; synchronous VGPR round-trip otherwise.
// Builtin signature (from hipcc diagnostic): (v4i addrspace(1)*, v4i addrspace(3)*, imm, imm)
// -------------------------------------------------------------------------
__device__ __forceinline__ void cp16_g2l(const __bf16* g, __bf16* l) {
#if __has_builtin(__builtin_amdgcn_global_load_async_to_lds_b128)
  __builtin_amdgcn_global_load_async_to_lds_b128(
      (__attribute__((address_space(1))) v4i*)g,
      (__attribute__((address_space(3))) v4i*)l, 0, 0);
#else
  *(float4*)l = *(const float4*)g;
#endif
}
__device__ __forceinline__ void g2l_wait() {
#if __has_builtin(__builtin_amdgcn_global_load_async_to_lds_b128)
#if __has_builtin(__builtin_amdgcn_s_wait_asynccnt)
  __builtin_amdgcn_s_wait_asynccnt(0);
#else
  asm volatile("s_wait_asynccnt 0x0" ::: "memory");
#endif
#endif
}

// -------------------------------------------------------------------------
// fp32 -> bf16 conversion (grid-stride)
// -------------------------------------------------------------------------
__global__ void f2bf_k(const float* __restrict__ s, __bf16* __restrict__ d, long long n) {
  long long i = (long long)blockIdx.x * blockDim.x + threadIdx.x;
  long long st = (long long)gridDim.x * blockDim.x;
  for (; i < n; i += st) d[i] = (__bf16)s[i];
}

// -------------------------------------------------------------------------
// prep: batch-major bf16 copies of tgt/memory, +pos variants, 1/|m| norms
// -------------------------------------------------------------------------
__global__ __launch_bounds__(256)
void prep_k(const float* __restrict__ tgt, const float* __restrict__ mem,
            const float* __restrict__ qpos, const float* __restrict__ kpos,
            __bf16* __restrict__ Xbf, __bf16* __restrict__ Mbf,
            __bf16* __restrict__ QCbf, __bf16* __restrict__ KCbf,
            float* __restrict__ minv) {
  int blk = blockIdx.x;
  int b = blk / MNP, m = blk % MNP, d = threadIdx.x;
  __shared__ float sh[256];
  float mv = mem[((long)m * BSZ + b) * DIM + d];
  long mo = ((long)b * MNP + m) * DIM + d;
  Mbf[mo]  = (__bf16)mv;
  KCbf[mo] = (__bf16)(mv + kpos[((long)m * BSZ + b) * DIM + d]);
  sh[d] = mv * mv; __syncthreads();
  for (int s = 128; s; s >>= 1) { if (d < s) sh[d] += sh[d + s]; __syncthreads(); }
  if (d == 0) minv[(long)b * MNP + m] = rsqrtf(fmaxf(sh[0], 1e-30f));
  if (m < NP) {
    float tv = tgt[((long)m * BSZ + b) * DIM + d];
    long xo = ((long)b * NP + m) * DIM + d;
    Xbf[xo]  = (__bf16)tv;
    QCbf[xo] = (__bf16)(tv + qpos[((long)m * BSZ + b) * DIM + d]);
  }
}

// -------------------------------------------------------------------------
// Generic batched WMMA bf16 GEMM:  C = act( scale * A*B(^T) + bias )
//   A: (M,K) row-major bf16.  BT=1: B is (N,K).  BT=0: B is (K,N).
//   OBF=1: bf16 output, else fp32.
//   TILE=1: 128x128 block, 8 waves each own 32x64 -> 8 WMMA / 32-k chunk
//   TILE=0: 64x64 block,  8 waves each own 16x32 -> 2 WMMA / 32-k chunk
// Global->LDS staging via async-to-LDS (ASYNCcnt) when available.
// Fragment layouts per CDNA5 ISA 7.12.2 (wave32).
// -------------------------------------------------------------------------
template<int BT, int OBF, int TILE>
__global__ __launch_bounds__(256)
void gemm_k(const __bf16* __restrict__ A, const __bf16* __restrict__ B,
            const float* __restrict__ bias, void* __restrict__ Cp,
            int M, int N, int K, int lda, int ldb, int ldc,
            long long sA1, long long sA2, long long sB1, long long sB2,
            long long sC1, long long sC2, int b2, float scale, int act) {
  constexpr int BM = TILE ? 128 : 64;
  constexpr int BN = TILE ? 128 : 64;
  constexpr int MW = TILE ? 2 : 1;     // 16-row subtiles per wave
  constexpr int NW = TILE ? 4 : 2;     // 16-col subtiles per wave
  constexpr int LP = 48;               // LDS pitch (bf16): 96B rows, 16B-aligned chunks
  constexpr int BSN = BT ? BN * LP : 32 * 72;
  __shared__ __bf16 As[BM * LP];
  __shared__ __bf16 Bs[BSN];
  int tid = threadIdx.x;
  int z = blockIdx.z, z1 = z / b2, z2 = z % b2;
  const __bf16* Ab = A + z1 * sA1 + z2 * sA2;
  const __bf16* Bb = B + z1 * sB1 + z2 * sB2;
  int mBase = blockIdx.x * BM, nBase = blockIdx.y * BN;
  int wid = tid >> 5, lane = tid & 31;
  int mr = (wid & 3) * (16 * MW), nc = (wid >> 2) * (16 * NW);
  int r = lane & 15, hh = lane >> 4;
  v8f acc[MW][NW] = {};

  for (int k0 = 0; k0 < K; k0 += 32) {
    // ---- stage A tile BMx32 (16B per thread per 64 rows) ----
#pragma unroll
    for (int t = 0; t < BM / 64; ++t) {
      int row = t * 64 + (tid >> 2), ko = (tid & 3) * 8;
      int gr = mBase + row; if (gr >= M) gr = M - 1;
      cp16_g2l(Ab + (long)gr * lda + k0 + ko, &As[row * LP + ko]);
    }
    if (BT) { // ---- B tile: BN n-rows x 32 k ----
#pragma unroll
      for (int t = 0; t < BN / 64; ++t) {
        int row = t * 64 + (tid >> 2), ko = (tid & 3) * 8;
        int gn = nBase + row; if (gn >= N) gn = N - 1;
        cp16_g2l(Bb + (long)gn * ldb + k0 + ko, &Bs[row * LP + ko]);
      }
    } else {  // ---- B tile: 32 k-rows x 64 n (strided gather) ----
      int kk = tid >> 3, no = (tid & 7) * 8;
#pragma unroll
      for (int i = 0; i < 8; ++i) {
        int gn = nBase + no + i; if (gn >= N) gn = N - 1;
        Bs[kk * 72 + no + i] = Bb[(long)(k0 + kk) * ldb + gn];
      }
    }
    if (k0 + 32 < K) // L2 prefetch of next A chunk
      __builtin_prefetch(Ab + (long)(mBase + (tid >> 2)) * lda + k0 + 32, 0, 0);
    g2l_wait();
    __syncthreads();

    // ---- fragments + WMMA ----
    v16bf af[MW];
#pragma unroll
    for (int mi = 0; mi < MW; ++mi)
#pragma unroll
      for (int e = 0; e < 16; ++e) {   // A 16x32: lane row r, K-half hh
        int kk = (e < 8) ? (8 * hh + e) : (8 * hh + e + 8);
        af[mi][e] = As[(mr + 16 * mi + r) * LP + kk];
      }
#pragma unroll
    for (int ni = 0; ni < NW; ++ni) {
      v16bf bfr;
#pragma unroll
      for (int e = 0; e < 16; ++e) {   // B 32x16: col r, K-half hh (same k order as A)
        int kk = (e < 8) ? (8 * hh + e) : (8 * hh + e + 8);
        bfr[e] = BT ? Bs[(nc + 16 * ni + r) * LP + kk]
                    : Bs[kk * 72 + nc + 16 * ni + r];
      }
#pragma unroll
      for (int mi = 0; mi < MW; ++mi)
        acc[mi][ni] = __builtin_amdgcn_wmma_f32_16x16x32_bf16(
            false, af[mi], false, bfr, (short)0, acc[mi][ni], false, false);
    }
    __syncthreads();
  }

  // ---- store: D element (m,n): vgpr j -> m = 8*hh + j, col = r ----
#pragma unroll
  for (int mi = 0; mi < MW; ++mi)
#pragma unroll
    for (int j = 0; j < 8; ++j) {
      int gm = mBase + mr + 16 * mi + 8 * hh + j;
      if (gm >= M) continue;
#pragma unroll
      for (int ni = 0; ni < NW; ++ni) {
        int gn = nBase + nc + 16 * ni + r;
        if (gn >= N) continue;
        float v = acc[mi][ni][j] * scale + (bias ? bias[gn] : 0.0f);
        if (act == 1) v = 0.5f * v * (1.0f + erff(v * 0.70710678118654752f));
        long o = (long)gm * ldc + gn + z1 * sC1 + z2 * sC2;
        if (OBF) ((__bf16*)Cp)[o] = (__bf16)v;
        else     ((float*)Cp)[o]  = v;
      }
    }
}

// -------------------------------------------------------------------------
// top-32 per row of sim (scaled by per-key 1/|m|); iterative block argmax.
// -------------------------------------------------------------------------
__global__ __launch_bounds__(256)
void topk_k(const float* __restrict__ sim, const float* __restrict__ minv,
            int* __restrict__ out) {
  int row = blockIdx.x, b = row / NP, tid = threadIdx.x;
  const float* sr = sim + (long)row * MNP;
  const float* mv = minv + (long)b * MNP;
  __shared__ float sv[256];
  __shared__ int   si[256];
  float v[16];
  int base = tid * 16;
#pragma unroll
  for (int i = 0; i < 16; ++i) v[i] = sr[base + i] * mv[base + i];
  unsigned alive = 0xFFFFu;
  for (int t = 0; t < TOPK; ++t) {
    float bv = -3.0e38f; int bi = 0x7FFFFFFF;
#pragma unroll
    for (int i = 0; i < 16; ++i)
      if ((alive >> i) & 1u) {
        float x = v[i];
        if (x > bv || (x == bv && base + i < bi)) { bv = x; bi = base + i; }
      }
    sv[tid] = bv; si[tid] = bi; __syncthreads();
    for (int s = 128; s; s >>= 1) {
      if (tid < s) {
        if (sv[tid + s] > sv[tid] || (sv[tid + s] == sv[tid] && si[tid + s] < si[tid])) {
          sv[tid] = sv[tid + s]; si[tid] = si[tid + s];
        }
      }
      __syncthreads();
    }
    int widx = si[0];
    if (tid == 0) out[(long)row * TOPK + t] = widx;
    if (widx < MNP && (widx >> 4) == tid) alive &= ~(1u << (widx & 15));
    __syncthreads();
  }
}

// -------------------------------------------------------------------------
// gathered cross-attention: one wave per (b, n, h); 32 top-k keys = 32 lanes.
// -------------------------------------------------------------------------
__global__ __launch_bounds__(256)
void cross_attn_k(const __bf16* __restrict__ qp, const __bf16* __restrict__ kp,
                  const __bf16* __restrict__ vp, const int* __restrict__ topk,
                  float* __restrict__ outc) {
  int g = blockIdx.x * 8 + (threadIdx.x >> 5);
  int lane = threadIdx.x & 31;
  int b = g / (NP * NH); int rem = g % (NP * NH);
  int n = rem / NH, h = rem % NH;
  long qoff = ((long)b * NP + n) * DIM + h * HD;
  float q = (float)qp[qoff + lane];
  int myidx = topk[((long)b * NP + n) * TOPK + lane];
  const float scale = 0.17677669529663687f; // 1/sqrt(32)
  float score = 0.0f;
#pragma unroll 4
  for (int j = 0; j < TOPK; ++j) {
    int idx = __shfl(myidx, j, 32);
    float kv = (float)kp[((long)b * MNP + idx) * DIM + h * HD + lane];
    float p = q * kv;
    for (int o = 16; o; o >>= 1) p += __shfl_xor(p, o, 32);
    if (lane == j) score = p * scale;
  }
  float mx = score;
  for (int o = 16; o; o >>= 1) mx = fmaxf(mx, __shfl_xor(mx, o, 32));
  float e = expf(score - mx), s = e;
  for (int o = 16; o; o >>= 1) s += __shfl_xor(s, o, 32);
  float w = e / s, out = 0.0f;
#pragma unroll 4
  for (int j = 0; j < TOPK; ++j) {
    float wj = __shfl(w, j, 32);
    int idx = __shfl(myidx, j, 32);
    out += wj * (float)vp[((long)b * MNP + idx) * DIM + h * HD + lane];
  }
  outc[qoff + lane] = out;
}

// -------------------------------------------------------------------------
// residual + LayerNorm (+ optional bf16 outputs and +pos bf16 output)
// -------------------------------------------------------------------------
__global__ __launch_bounds__(256)
void add_ln_k(const float* __restrict__ resid, int residT,
              const float* __restrict__ x,
              const float* __restrict__ w, const float* __restrict__ bsc,
              float* __restrict__ outf, int outT,
              __bf16* __restrict__ outbf,
              const float* __restrict__ pos, __bf16* __restrict__ outposbf) {
  int b = blockIdx.x / NP, n = blockIdx.x % NP, d = threadIdx.x;
  __shared__ float sh[256];
  float rv = residT ? resid[((long)n * BSZ + b) * DIM + d]
                    : resid[((long)b * NP + n) * DIM + d];
  float v = rv + x[((long)b * NP + n) * DIM + d];
  sh[d] = v; __syncthreads();
  for (int s = 128; s; s >>= 1) { if (d < s) sh[d] += sh[d + s]; __syncthreads(); }
  float mu = sh[0] * (1.0f / DIM); __syncthreads();
  float t = v - mu;
  sh[d] = t * t; __syncthreads();
  for (int s = 128; s; s >>= 1) { if (d < s) sh[d] += sh[d + s]; __syncthreads(); }
  float var = sh[0] * (1.0f / DIM);
  float y = t * rsqrtf(var + 1e-5f) * w[d] + bsc[d];
  if (outf) {
    long o = outT ? ((long)n * BSZ + b) * DIM + d : ((long)b * NP + n) * DIM + d;
    outf[o] = y;
  }
  if (outbf)    outbf[((long)b * NP + n) * DIM + d] = (__bf16)y;
  if (outposbf) outposbf[((long)b * NP + n) * DIM + d] =
                  (__bf16)(y + pos[((long)n * BSZ + b) * DIM + d]);
}

// -------------------------------------------------------------------------
// row softmax over 512 cols, fp32 in -> bf16 probs. One wave per row.
// -------------------------------------------------------------------------
__global__ __launch_bounds__(256)
void softmax_k(const float* __restrict__ S, __bf16* __restrict__ P) {
  int row = blockIdx.x * 8 + (threadIdx.x >> 5);
  int lane = threadIdx.x & 31;
  const float* sr = S + (long)row * NP;
  float vals[16]; float mx = -3.0e38f;
#pragma unroll
  for (int i = 0; i < 16; ++i) { vals[i] = sr[lane + i * 32]; mx = fmaxf(mx, vals[i]); }
  for (int o = 16; o; o >>= 1) mx = fmaxf(mx, __shfl_xor(mx, o, 32));
  float s = 0.0f;
#pragma unroll
  for (int i = 0; i < 16; ++i) { vals[i] = expf(vals[i] - mx); s += vals[i]; }
  for (int o = 16; o; o >>= 1) s += __shfl_xor(s, o, 32);
  float inv = 1.0f / s;
#pragma unroll
  for (int i = 0; i < 16; ++i) P[(long)row * NP + lane + i * 32] = (__bf16)(vals[i] * inv);
}

// -------------------------------------------------------------------------
extern "C" void kernel_launch(void* const* d_in, const int* in_sizes, int n_in,
                              void* d_out, int out_size, void* d_ws, size_t ws_size,
                              hipStream_t stream) {
  const float* tgt  = (const float*)d_in[0];
  const float* mem  = (const float*)d_in[1];
  const float* qpos = (const float*)d_in[2];
  const float* kpos = (const float*)d_in[3];
  const float* cw1 = (const float*)d_in[4];  const float* cb1 = (const float*)d_in[5];
  const float* cw2 = (const float*)d_in[6];  const float* cb2 = (const float*)d_in[7];
  const float* cw3 = (const float*)d_in[8];  const float* cb3 = (const float*)d_in[9];
  const float* saw = (const float*)d_in[10]; const float* sab = (const float*)d_in[11];
  const float* sow = (const float*)d_in[12]; const float* sob = (const float*)d_in[13];
  const float* ln1w = (const float*)d_in[14]; const float* ln1b = (const float*)d_in[15];
  const float* ln2w = (const float*)d_in[16]; const float* ln2b = (const float*)d_in[17];
  const float* ln3w = (const float*)d_in[18]; const float* ln3b = (const float*)d_in[19];
  const float* f1w = (const float*)d_in[20]; const float* f1b = (const float*)d_in[21];
  const float* f2w = (const float*)d_in[22]; const float* f2b = (const float*)d_in[23];

  char* p = (char*)d_ws;
  auto take = [&](size_t bytes) -> void* {
    void* r = (void*)p; p += (bytes + 255) & ~(size_t)255; return r;
  };
  __bf16* Xbf   = (__bf16*)take((size_t)BSZ * NP  * DIM * 2);
  __bf16* Mbf   = (__bf16*)take((size_t)BSZ * MNP * DIM * 2);
  __bf16* QCbf  = (__bf16*)take((size_t)BSZ * NP  * DIM * 2);
  __bf16* KCbf  = (__bf16*)take((size_t)BSZ * MNP * DIM * 2);
  float*  minv  = (float*) take((size_t)BSZ * MNP * 4);
  float*  simS  = (float*) take((size_t)BSZ * NP * MNP * 4);   // reused as self-attn scores
  int*    tki   = (int*)   take((size_t)BSZ * NP * TOPK * 4);
  __bf16* qprj  = (__bf16*)take((size_t)BSZ * NP  * DIM * 2);
  __bf16* kprj  = (__bf16*)take((size_t)BSZ * MNP * DIM * 2);
  __bf16* vprj  = (__bf16*)take((size_t)BSZ * MNP * DIM * 2);
  float*  crossf= (float*) take((size_t)BSZ * NP * DIM * 4);
  float*  t1f   = (float*) take((size_t)BSZ * NP * DIM * 4);
  __bf16* t1bf  = (__bf16*)take((size_t)BSZ * NP * DIM * 2);
  __bf16* qpbf  = (__bf16*)take((size_t)BSZ * NP * DIM * 2);
  __bf16* q2bf  = (__bf16*)take((size_t)BSZ * NP * DIM * 2);
  __bf16* k2bf  = (__bf16*)take((size_t)BSZ * NP * DIM * 2);
  __bf16* v2bf  = (__bf16*)take((size_t)BSZ * NP * DIM * 2);
  __bf16* Pbf   = (__bf16*)take((size_t)BSZ * NH * NP * NP * 2);
  __bf16* sabf  = (__bf16*)take((size_t)BSZ * NP * DIM * 2);
  float*  saof  = (float*) take((size_t)BSZ * NP * DIM * 4);
  float*  t2f   = (float*) take((size_t)BSZ * NP * DIM * 4);
  __bf16* t2bf  = (__bf16*)take((size_t)BSZ * NP * DIM * 2);
  __bf16* hbf   = (__bf16*)take((size_t)BSZ * NP * FF * 2);
  float*  fff   = (float*) take((size_t)BSZ * NP * DIM * 4);
  __bf16* cw1b  = (__bf16*)take((size_t)DIM * DIM * 2);
  __bf16* cw2b  = (__bf16*)take((size_t)DIM * DIM * 2);
  __bf16* cw3b  = (__bf16*)take((size_t)DIM * DIM * 2);
  __bf16* sawb  = (__bf16*)take((size_t)3 * DIM * DIM * 2);
  __bf16* sowb  = (__bf16*)take((size_t)DIM * DIM * 2);
  __bf16* f1wb  = (__bf16*)take((size_t)FF * DIM * 2);
  __bf16* f2wb  = (__bf16*)take((size_t)DIM * FF * 2);

  dim3 blk(256);
  const float iscale = 0.17677669529663687f; // 1/sqrt(HD)

  // weight conversions
  f2bf_k<<<256, blk, 0, stream>>>(cw1, cw1b, (long long)DIM * DIM);
  f2bf_k<<<256, blk, 0, stream>>>(cw2, cw2b, (long long)DIM * DIM);
  f2bf_k<<<256, blk, 0, stream>>>(cw3, cw3b, (long long)DIM * DIM);
  f2bf_k<<<256, blk, 0, stream>>>(saw, sawb, (long long)3 * DIM * DIM);
  f2bf_k<<<256, blk, 0, stream>>>(sow, sowb, (long long)DIM * DIM);
  f2bf_k<<<256, blk, 0, stream>>>(f1w, f1wb, (long long)FF * DIM);
  f2bf_k<<<256, blk, 0, stream>>>(f2w, f2wb, (long long)DIM * FF);

  // prep: transposes, +pos, key norms
  prep_k<<<BSZ * MNP, blk, 0, stream>>>(tgt, mem, qpos, kpos, Xbf, Mbf, QCbf, KCbf, minv);

  auto g128 = [](int M, int N, int Z) { return dim3((M + 127) / 128, (N + 127) / 128, Z); };
  auto g64  = [](int M, int N, int Z) { return dim3((M + 63) / 64, (N + 63) / 64, Z); };

  // sim = X @ M^T (per batch), fp32
  gemm_k<1,0,1><<<g128(NP, MNP, BSZ), blk, 0, stream>>>(
      Xbf, Mbf, nullptr, simS, NP, MNP, DIM, DIM, DIM, MNP,
      (long long)NP * DIM, 0, (long long)MNP * DIM, 0, (long long)NP * MNP, 0,
      1, 1.0f, 0);
  topk_k<<<BSZ * NP, blk, 0, stream>>>(simS, minv, tki);

  // cross-attn projections (bf16 out)
  gemm_k<1,1,1><<<g128(NP, DIM, BSZ), blk, 0, stream>>>(
      QCbf, cw1b, cb1, qprj, NP, DIM, DIM, DIM, DIM, DIM,
      (long long)NP * DIM, 0, 0, 0, (long long)NP * DIM, 0, 1, 1.0f, 0);
  gemm_k<1,1,1><<<g128(MNP, DIM, BSZ), blk, 0, stream>>>(
      KCbf, cw2b, cb2, kprj, MNP, DIM, DIM, DIM, DIM, DIM,
      (long long)MNP * DIM, 0, 0, 0, (long long)MNP * DIM, 0, 1, 1.0f, 0);
  gemm_k<1,1,1><<<g128(MNP, DIM, BSZ), blk, 0, stream>>>(
      Mbf, cw3b, cb3, vprj, MNP, DIM, DIM, DIM, DIM, DIM,
      (long long)MNP * DIM, 0, 0, 0, (long long)MNP * DIM, 0, 1, 1.0f, 0);

  // gathered top-32 cross attention
  cross_attn_k<<<(BSZ * NP * NH) / 8, blk, 0, stream>>>(qprj, kprj, vprj, tki, crossf);

  // LN1 (+ t1 bf16, + (t1+qpos) bf16)
  add_ln_k<<<BSZ * NP, blk, 0, stream>>>(tgt, 1, crossf, ln1w, ln1b, t1f, 0, t1bf, qpos, qpbf);

  // self-attn projections
  gemm_k<1,1,1><<<g128(NP, DIM, BSZ), blk, 0, stream>>>(
      qpbf, sawb, sab, q2bf, NP, DIM, DIM, DIM, DIM, DIM,
      (long long)NP * DIM, 0, 0, 0, (long long)NP * DIM, 0, 1, 1.0f, 0);
  gemm_k<1,1,1><<<g128(NP, DIM, BSZ), blk, 0, stream>>>(
      qpbf, sawb + DIM * DIM, sab + DIM, k2bf, NP, DIM, DIM, DIM, DIM, DIM,
      (long long)NP * DIM, 0, 0, 0, (long long)NP * DIM, 0, 1, 1.0f, 0);
  gemm_k<1,1,1><<<g128(NP, DIM, BSZ), blk, 0, stream>>>(
      t1bf, sawb + 2 * DIM * DIM, sab + 2 * DIM, v2bf, NP, DIM, DIM, DIM, DIM, DIM,
      (long long)NP * DIM, 0, 0, 0, (long long)NP * DIM, 0, 1, 1.0f, 0);

  // scores S[b,h] = Q2 @ K2^T * scale (reuse simS)
  gemm_k<1,0,1><<<g128(NP, NP, BSZ * NH), blk, 0, stream>>>(
      q2bf, k2bf, nullptr, simS, NP, NP, HD, DIM, DIM, NP,
      (long long)NP * DIM, HD, (long long)NP * DIM, HD,
      (long long)NH * NP * NP, (long long)NP * NP, NH, iscale, 0);

  softmax_k<<<(BSZ * NH * NP) / 8, blk, 0, stream>>>(simS, Pbf);

  // SA[b,h] = P @ V2 (NN) -> packed straight into (b, n, h*32+d) bf16
  gemm_k<0,1,0><<<g64(NP, HD, BSZ * NH), blk, 0, stream>>>(
      Pbf, v2bf, nullptr, sabf, NP, HD, NP, NP, DIM, DIM,
      (long long)NH * NP * NP, (long long)NP * NP, (long long)NP * DIM, HD,
      (long long)NP * DIM, HD, NH, 1.0f, 0);

  // output projection
  gemm_k<1,0,1><<<g128(NP, DIM, BSZ), blk, 0, stream>>>(
      sabf, sowb, sob, saof, NP, DIM, DIM, DIM, DIM, DIM,
      (long long)NP * DIM, 0, 0, 0, (long long)NP * DIM, 0, 1, 1.0f, 0);

  // LN2
  add_ln_k<<<BSZ * NP, blk, 0, stream>>>(t1f, 0, saof, ln2w, ln2b, t2f, 0, t2bf, nullptr, nullptr);

  // FFN
  gemm_k<1,1,1><<<g128(NP, FF, BSZ), blk, 0, stream>>>(
      t2bf, f1wb, f1b, hbf, NP, FF, DIM, DIM, DIM, FF,
      (long long)NP * DIM, 0, 0, 0, (long long)NP * FF, 0, 1, 1.0f, 1 /*GELU*/);
  gemm_k<1,0,1><<<g128(NP, DIM, BSZ), blk, 0, stream>>>(
      hbf, f2wb, f2b, fff, NP, DIM, FF, FF, FF, DIM,
      (long long)NP * FF, 0, 0, 0, (long long)NP * DIM, 0, 1, 1.0f, 0);

  // LN3 -> d_out in (np, bs, dim) layout
  add_ln_k<<<BSZ * NP, blk, 0, stream>>>(t2f, 0, fff, ln3w, ln3b, (float*)d_out, 1,
                                         nullptr, nullptr, nullptr);
  (void)in_sizes; (void)n_in; (void)out_size; (void)ws_size;
}
